// MinGRU_23210003268343
// MI455X (gfx1250) — compile-verified, matrix operand order checked by reading, not run on hardware
//
#include <hip/hip_runtime.h>
#include <math.h>

// ---------------------------------------------------------------------------
// MinGRU (2-layer bidirectional) for MI455X / gfx1250, wave32 + WMMA bf16.
//
//   1) convert x / weights to bf16 (weights*0.05, activations O(1) -> bf16
//      A/B with f32 WMMA accumulate is the precision the chip rewards).
//   2) GEMM hg = cur @ W^T via v_wmma_f32_16x16x32_bf16:
//        - B panel (128 cols x K) staged in LDS once per block, padded to
//          kill bank conflicts; persisted across 4 M-tiles per block.
//        - A fragments double-buffered in registers (global loads for k+32
//          issued before the 4 WMMAs of k), so HBM latency overlaps WMMA.
//   3) chunked linear scan h_t = a*h + b (T=8192 -> 64 chunks of 128):
//      pass1 per-chunk summaries (131072 threads), pass2 carry sweep
//      (2048 threads, also emits last-states), pass3 replay+write.
// ---------------------------------------------------------------------------

typedef __attribute__((ext_vector_type(16))) __bf16 v16bf;
typedef __attribute__((ext_vector_type(8)))  __bf16 v8bf;
typedef __attribute__((ext_vector_type(8)))  float  v8f;

#define B_    8
#define T_    8192
#define D_    256
#define CH_   256            // H (channels per direction)
#define NH2_  512            // 2H
#define M_    (B_ * T_)      // 65536 GEMM rows
#define L_    128            // scan chunk length
#define NC_   (T_ / L_)      // 64 chunks
#define NSEQ_ (B_ * CH_)     // 2048 independent sequences per direction
#define MT_   4              // M tiles (of 64 rows) per block in GEMM

// ---------------------------------------------------------------- conversions
__global__ __launch_bounds__(256) void f32_to_bf16_kernel(
    const float* __restrict__ in, __bf16* __restrict__ out, int n) {
  int i = blockIdx.x * blockDim.x + threadIdx.x;
  if (i < n) out[i] = (__bf16)in[i];
}

// ---------------------------------------------------------------- GEMM (WMMA)
// Out[M x N] = A[M x K] * W^T, W stored N x K row-major (B column-major =
// per-lane layout symmetric to A).
// Block: 256 threads = 8 waves; N panel = 128 cols staged in LDS; block
// sweeps MT_ tiles of 64 M rows; wave tile 16M x 64N (4 accumulators).
template <int K>
__global__ __launch_bounds__(256) void gemm_bf16_wmma(
    const __bf16* __restrict__ A,   // M x K row-major
    const __bf16* __restrict__ W,   // N x K row-major
    float* __restrict__ out,        // M x N row-major
    int N)
{
  constexpr int STRIDE = K + 8;     // +8 bf16 = +16B row pad -> conflict-free ds_load_b128
  extern __shared__ char smem_raw[];
  __bf16* Bs = (__bf16*)smem_raw;   // 128 x STRIDE bf16

  const int ncolBase = blockIdx.y * 128;

  // ---- stage B panel (128 rows x K) into LDS, 16B chunks, coalesced ----
  for (int idx = threadIdx.x; idx < 128 * (K / 8); idx += 256) {
    int r  = idx / (K / 8);
    int ck = idx - r * (K / 8);
    *(v8bf*)(&Bs[r * STRIDE + ck * 8]) =
        *(const v8bf*)(W + (size_t)(ncolBase + r) * K + ck * 8);
  }
  __syncthreads();

  const int lane  = threadIdx.x & 31;
  const int w     = threadIdx.x >> 5;   // 0..7
  const int wm    = w >> 1;             // 0..3
  const int wn    = w & 1;              // 0..1
  const int l16   = lane & 15;
  const int khalf = (lane >> 4) << 3;   // lanes 0-15: K 0..7/16..23; 16-31: 8..15/24..31

  // LDS base for this wave's B columns (wn*64 .. +63)
  const __bf16* bs = &Bs[(wn * 64 + l16) * STRIDE + khalf];

  union AF { v16bf v; v8bf h[2]; };

  for (int mt = 0; mt < MT_; ++mt) {
    const int mrow = (blockIdx.x * MT_ + mt) * 64 + wm * 16;
    const __bf16* aRow = A + (size_t)(mrow + l16) * K + khalf;

    v8f acc[4];
#pragma unroll
    for (int j = 0; j < 4; ++j) acc[j] = (v8f)(0.0f);

    AF cur, nxt;
    cur.h[0] = *(const v8bf*)(aRow);
    cur.h[1] = *(const v8bf*)(aRow + 16);

#pragma unroll
    for (int k0 = 0; k0 < K; k0 += 32) {
      // prefetch next A fragment first: global latency hides under 4 WMMAs
      if (k0 + 32 < K) {
        nxt.h[0] = *(const v8bf*)(aRow + k0 + 32);
        nxt.h[1] = *(const v8bf*)(aRow + k0 + 48);
      }
      // batch all B fragment ds_loads, then the 4 WMMAs
      union { v16bf v; v8bf h[2]; } bf[4];
#pragma unroll
      for (int j = 0; j < 4; ++j) {
        bf[j].h[0] = *(const v8bf*)(bs + (j * 16) * STRIDE + k0);
        bf[j].h[1] = *(const v8bf*)(bs + (j * 16) * STRIDE + k0 + 16);
      }
#pragma unroll
      for (int j = 0; j < 4; ++j) {
        acc[j] = __builtin_amdgcn_wmma_f32_16x16x32_bf16(
            false, cur.v, false, bf[j].v, (short)0, acc[j], false, false);
      }
      cur = nxt;
    }

    // C/D layout: VGPR r -> (M = r [lanes 0-15] | 8+r [lanes 16-31], N = lane&15)
    const int rbase = mrow + ((lane >> 4) << 3);
    const int ncol  = ncolBase + wn * 64;
#pragma unroll
    for (int j = 0; j < 4; ++j) {
      const int col = ncol + j * 16 + l16;
#pragma unroll
      for (int r = 0; r < 8; ++r) {
        out[(size_t)(rbase + r) * N + col] = acc[j][r];
      }
    }
  }
}

// ------------------------------------------------------------------ scan math
// a = sigmoid(-gate) = exp(-softplus(gate))
// b = sigmoid(gate) * g(h),  g(h) = h + 0.5 (h>=0) | sigmoid(h) (h<0)
__device__ __forceinline__ void coeffs(float h, float g, float& a, float& bv) {
  float sig = 1.0f / (1.0f + __expf(-g));      // sigmoid(gate)
  a = 1.0f / (1.0f + __expf(g));               // sigmoid(-gate)
  float gh = (h >= 0.0f) ? (h + 0.5f) : (1.0f / (1.0f + __expf(-h)));
  bv = sig * gh;
}

// tid layout for pass1/pass3: c = tid&255 (coalesced), k = (tid>>8)&63, b = tid>>14
__global__ __launch_bounds__(256) void scan_pass1(
    const float* __restrict__ hg,    // (B*T) x 512, h at col c, gate at col 256+c
    float* __restrict__ Abuf, float* __restrict__ Hbuf, int rev)
{
  int tid = blockIdx.x * 256 + threadIdx.x;
  int c = tid & 255;
  int k = (tid >> 8) & (NC_ - 1);
  int b = tid >> 14;
  float A = 1.0f, Hv = 0.0f;
  for (int i = 0; i < L_; ++i) {
    int tt = k * L_ + i;
    int t  = rev ? (T_ - 1 - tt) : tt;
    size_t row = (size_t)(b * T_ + t) * NH2_;
    float h = hg[row + c];
    float g = hg[row + CH_ + c];
    float a, bv; coeffs(h, g, a, bv);
    A  *= a;
    Hv  = a * Hv + bv;
  }
  int seq = b * CH_ + c;
  Abuf[(size_t)k * NSEQ_ + seq] = A;
  Hbuf[(size_t)k * NSEQ_ + seq] = Hv;
}

__global__ __launch_bounds__(256) void scan_pass2(
    const float* __restrict__ Abuf, const float* __restrict__ Hbuf,
    float* __restrict__ Cin, float* __restrict__ lastOut /* B*CH floats */)
{
  int tid = blockIdx.x * 256 + threadIdx.x;   // 0..NSEQ_-1 == b*256 + c
  if (tid >= NSEQ_) return;
  float carry = 0.0f;
  for (int k = 0; k < NC_; ++k) {
    Cin[(size_t)k * NSEQ_ + tid] = carry;
    carry = Abuf[(size_t)k * NSEQ_ + tid] * carry + Hbuf[(size_t)k * NSEQ_ + tid];
  }
  lastOut[tid] = carry;   // == reference's out[:, -1] (pre-unreverse for bwd)
}

template <typename OUT_T>
__global__ __launch_bounds__(256) void scan_pass3(
    const float* __restrict__ hg, const float* __restrict__ Cin,
    OUT_T* __restrict__ outCur,   // (B*T) x 512
    int rev, int colOff)
{
  int tid = blockIdx.x * 256 + threadIdx.x;
  int c = tid & 255;
  int k = (tid >> 8) & (NC_ - 1);
  int b = tid >> 14;
  int seq = b * CH_ + c;
  float hstate = Cin[(size_t)k * NSEQ_ + seq];
  for (int i = 0; i < L_; ++i) {
    int tt = k * L_ + i;
    int t  = rev ? (T_ - 1 - tt) : tt;
    size_t row = (size_t)(b * T_ + t) * NH2_;
    float h = hg[row + c];
    float g = hg[row + CH_ + c];
    float a, bv; coeffs(h, g, a, bv);
    hstate = a * hstate + bv;
    outCur[row + colOff + c] = (OUT_T)hstate;
  }
}

// ---------------------------------------------------------------------------
extern "C" void kernel_launch(void* const* d_in, const int* in_sizes, int n_in,
                              void* d_out, int out_size, void* d_ws, size_t ws_size,
                              hipStream_t stream) {
  const float* x   = (const float*)d_in[0];   // (8, 8192, 256)
  const float* W0f = (const float*)d_in[1];   // (512, 256)
  const float* W0b = (const float*)d_in[2];   // (512, 256)
  const float* W1f = (const float*)d_in[3];   // (512, 512)
  const float* W1b = (const float*)d_in[4];   // (512, 512)
  float* out = (float*)d_out;                 // cur (B,T,512) then h (4,B,256)

  // ---- workspace layout (256B aligned slices) ----
  char* ws = (char*)d_ws;
  size_t off = 0;
  auto carve = [&](size_t bytes) -> char* {
    char* p = ws + off;
    off = (off + bytes + 255) & ~(size_t)255;
    return p;
  };
  __bf16* xbf    = (__bf16*)carve((size_t)M_ * D_ * 2);
  __bf16* w0fbf  = (__bf16*)carve((size_t)NH2_ * D_ * 2);
  __bf16* w0bbf  = (__bf16*)carve((size_t)NH2_ * D_ * 2);
  __bf16* w1fbf  = (__bf16*)carve((size_t)NH2_ * NH2_ * 2);
  __bf16* w1bbf  = (__bf16*)carve((size_t)NH2_ * NH2_ * 2);
  __bf16* curbf  = (__bf16*)carve((size_t)M_ * NH2_ * 2);
  float*  hgF    = (float*)carve((size_t)M_ * NH2_ * 4);
  float*  hgB    = (float*)carve((size_t)M_ * NH2_ * 4);
  float*  Abuf   = (float*)carve((size_t)NC_ * NSEQ_ * 4);
  float*  Hbuf   = (float*)carve((size_t)NC_ * NSEQ_ * 4);
  float*  Cin    = (float*)carve((size_t)NC_ * NSEQ_ * 4);
  (void)ws_size; (void)n_in; (void)in_sizes; (void)out_size;

  const size_t CUR_ELEMS = (size_t)M_ * NH2_;
  float* hOut = out + CUR_ELEMS;   // (4, B, 256): [h0f, h0b, h1f, h1b]

  // ---- 1) f32 -> bf16 conversions ----
  {
    int n = M_ * D_;
    f32_to_bf16_kernel<<<(n + 255) / 256, 256, 0, stream>>>(x, xbf, n);
    n = NH2_ * D_;
    f32_to_bf16_kernel<<<(n + 255) / 256, 256, 0, stream>>>(W0f, w0fbf, n);
    f32_to_bf16_kernel<<<(n + 255) / 256, 256, 0, stream>>>(W0b, w0bbf, n);
    n = NH2_ * NH2_;
    f32_to_bf16_kernel<<<(n + 255) / 256, 256, 0, stream>>>(W1f, w1fbf, n);
    f32_to_bf16_kernel<<<(n + 255) / 256, 256, 0, stream>>>(W1b, w1bbf, n);
  }

  dim3 gemmGrid(M_ / (64 * MT_), NH2_ / 128);     // (256, 4)
  const size_t lds256 = 128 * (256 + 8) * 2;      // 67.6 KB
  const size_t lds512 = 128 * (512 + 8) * 2;      // 133 KB (<= 320 KB WGP LDS)
  dim3 scanGrid((NSEQ_ * NC_) / 256);
  dim3 carryGrid(NSEQ_ / 256);

  // ---- 2) layer 0 GEMMs: hg = x @ W0{f,b}^T (K = 256) ----
  gemm_bf16_wmma<256><<<gemmGrid, 256, lds256, stream>>>(xbf, w0fbf, hgF, NH2_);
  gemm_bf16_wmma<256><<<gemmGrid, 256, lds256, stream>>>(xbf, w0bbf, hgB, NH2_);

  // ---- 3) layer 0 scans -> curbf (bf16), last states -> hOut[0..1] ----
  scan_pass1<<<scanGrid, 256, 0, stream>>>(hgF, Abuf, Hbuf, /*rev=*/0);
  scan_pass2<<<carryGrid, 256, 0, stream>>>(Abuf, Hbuf, Cin, hOut + 0 * NSEQ_);
  scan_pass3<__bf16><<<scanGrid, 256, 0, stream>>>(hgF, Cin, curbf, 0, /*colOff=*/0);

  scan_pass1<<<scanGrid, 256, 0, stream>>>(hgB, Abuf, Hbuf, /*rev=*/1);
  scan_pass2<<<carryGrid, 256, 0, stream>>>(Abuf, Hbuf, Cin, hOut + 1 * NSEQ_);
  scan_pass3<__bf16><<<scanGrid, 256, 0, stream>>>(hgB, Cin, curbf, 1, /*colOff=*/CH_);

  // ---- 4) layer 1 GEMMs: hg = cur @ W1{f,b}^T (K = 512) ----
  gemm_bf16_wmma<512><<<gemmGrid, 256, lds512, stream>>>(curbf, w1fbf, hgF, NH2_);
  gemm_bf16_wmma<512><<<gemmGrid, 256, lds512, stream>>>(curbf, w1bbf, hgB, NH2_);

  // ---- 5) layer 1 scans -> d_out (f32), last states -> hOut[2..3] ----
  scan_pass1<<<scanGrid, 256, 0, stream>>>(hgF, Abuf, Hbuf, /*rev=*/0);
  scan_pass2<<<carryGrid, 256, 0, stream>>>(Abuf, Hbuf, Cin, hOut + 2 * NSEQ_);
  scan_pass3<float><<<scanGrid, 256, 0, stream>>>(hgF, Cin, out, 0, /*colOff=*/0);

  scan_pass1<<<scanGrid, 256, 0, stream>>>(hgB, Abuf, Hbuf, /*rev=*/1);
  scan_pass2<<<carryGrid, 256, 0, stream>>>(Abuf, Hbuf, Cin, hOut + 3 * NSEQ_);
  scan_pass3<float><<<scanGrid, 256, 0, stream>>>(hgB, Cin, out, 1, /*colOff=*/CH_);
}
